// PermutohedralLattice_58471684768093
// MI455X (gfx1250) — matrix operand order repaired and checked by reading, not compiled, not run on hardware
//
#include <hip/hip_runtime.h>

typedef __attribute__((ext_vector_type(16))) _Float16 v16h;
typedef __attribute__((ext_vector_type(8)))  float    v8f;

#define N_PTS   16384
#define DF      64
#define NNB     15
#define CAP     131072            // >= 2x max distinct keys (65536) -> load factor <= 0.5
#define HMASK   (CAP - 1)
#define KEMPTY  0xFFFFFFFFu

// First-3 coords of the reference's _neighbor_offsets(3,1), zero offset first.
__constant__ int c_off[NNB][3] = {
  { 0, 0, 0},{ 1, 1, 1},{ 1, 1,-3},{ 2, 2,-2},{ 1,-3, 1},
  { 2,-2, 2},{ 2,-2,-2},{ 3,-1,-1},{-3, 1, 1},{-2, 2, 2},
  {-2, 2,-2},{-1, 3,-1},{-2,-2, 2},{-1,-1, 3},{-1,-1,-1}
};

// ---------------------------------------------------------------- lattice ---
__device__ __forceinline__ void lattice3(float px, float py, float pz,
                                         int rem0[4], int rank[4], float bary[4])
{
  // pos @ EMAT for d=3 (exact column coefficients of the elevation matrix)
  float elev[4];
  elev[0] =  2.3094010767585034f*px + 1.3333333333333333f*py + 0.9428090415820634f*pz;
  elev[1] = -2.3094010767585034f*px + 1.3333333333333333f*py + 0.9428090415820634f*pz;
  elev[2] =                          -2.6666666666666667f*py + 0.9428090415820634f*pz;
  elev[3] =                                                   -2.8284271247461903f*pz;
  float diff[4]; int sum = 0;
  #pragma unroll
  for (int i = 0; i < 4; ++i) {
    float r = rintf(elev[i] * 0.25f) * 4.0f;   // round-half-even == jnp.round
    rem0[i] = (int)r;
    diff[i] = elev[i] - r;
    sum += rem0[i];
  }
  int s = sum >> 2;                            // sum is a multiple of 4
  #pragma unroll
  for (int i = 0; i < 4; ++i) {
    int rk = 0;                                // stable double-argsort rank
    #pragma unroll
    for (int j = 0; j < 4; ++j)
      rk += (diff[j] > diff[i]) || (diff[j] == diff[i] && j < i);
    rk += s;
    if (rk < 0)      { rk += 4; rem0[i] += 4; }
    else if (rk > 3) { rk -= 4; rem0[i] -= 4; }
    rank[i] = rk;
  }
  float b[5] = {0.f, 0.f, 0.f, 0.f, 0.f};
  #pragma unroll
  for (int i = 0; i < 4; ++i) {
    float t = (elev[i] - (float)rem0[i]) * 0.25f;
    b[3 - rank[i]] += t;
    b[4 - rank[i]] -= t;
  }
  b[0] += 1.0f + b[4];
  bary[0] = b[0]; bary[1] = b[1]; bary[2] = b[2]; bary[3] = b[3];
}

// ------------------------------------------------------------------- hash ---
__device__ __forceinline__ unsigned packKey(int k0, int k1, int k2) {
  return (unsigned)(k0 + 512) | ((unsigned)(k1 + 512) << 10) | ((unsigned)(k2 + 512) << 20);
}
__device__ __forceinline__ unsigned hashKey(unsigned p) {
  p *= 0x9E3779B1u; p ^= p >> 16; return p & HMASK;
}
__device__ int hashInsert(unsigned* kt, unsigned pk) {
  unsigned h = hashKey(pk);
  for (int probe = 0; probe < CAP; ++probe) {
    unsigned prev = atomicCAS(&kt[h], KEMPTY, pk);
    if (prev == KEMPTY || prev == pk) return (int)h;
    h = (h + 1) & HMASK;
  }
  return -1;
}
__device__ int hashFind(const unsigned* kt, unsigned pk) {
  unsigned h = hashKey(pk);
  for (int probe = 0; probe < CAP; ++probe) {
    unsigned k = kt[h];
    if (k == pk)    return (int)h;
    if (k == KEMPTY) return -1;
    h = (h + 1) & HMASK;
  }
  return -1;
}

// ------------------------------------------------------------------ splat ---
__global__ void pl_splat(const float* __restrict__ feat,    // [B,64,N]
                         const float* __restrict__ posIn,   // [B,N,3]
                         unsigned*    __restrict__ keysT,   // [B,CAP]
                         float*       __restrict__ barysum, // [B,CAP]
                         float*       __restrict__ vals)    // [B,CAP,64]
{
  const int p = blockIdx.x * blockDim.x + threadIdx.x;
  const int b = blockIdx.y;
  if (p >= N_PTS) return;
  const float* pp = posIn + ((size_t)b * N_PTS + p) * 3;
  int rem0[4], rank[4]; float bary[4];
  lattice3(pp[0]*4.f, pp[1]*4.f, pp[2]*4.f, rem0, rank, bary);

  unsigned* kt = keysT   + (size_t)b * CAP;
  float*    bs = barysum + (size_t)b * CAP;
  float*    vv = vals    + (size_t)b * CAP * DF;
  const float* fb = feat + (size_t)b * DF * N_PTS + p;

  #pragma unroll
  for (int j = 0; j < 4; ++j) {
    int k0 = rem0[0] + (rank[0] < 4 - j ? j : j - 4);
    int k1 = rem0[1] + (rank[1] < 4 - j ? j : j - 4);
    int k2 = rem0[2] + (rank[2] < 4 - j ? j : j - 4);
    int slot = hashInsert(kt, packKey(k0, k1, k2));
    if (slot < 0) continue;
    float w = bary[j];
    atomicAdd(&bs[slot], w);
    float* vrow = vv + (size_t)slot * DF;
    for (int c = 0; c < DF; ++c)
      atomicAdd(&vrow[c], w * fb[(size_t)c * N_PTS]);
  }
}

// ------------------------- weights -> per-lane WMMA B-fragment order -------
// Fragment (n, c, t): B[k][j] = W[o=16t+j][i=32c+k][n], lane L holds
// N-col j=L&15, K rows k=(L<16?0:8)+h+(h<8?0:8), h=0..15.  Stored so each lane's 16
// halves are contiguous (32B):  Wf[((n*2+c)*4+t)*512 + L*16 + h]
__global__ void pl_wswizzle(const float* __restrict__ W, _Float16* __restrict__ Wf)
{
  int idx = blockIdx.x * blockDim.x + threadIdx.x;   // 15*2*4*32*16 = 61440
  if (idx >= NNB * 2 * 4 * 32 * 16) return;
  int h = idx & 15;
  int L = (idx >> 4) & 31;
  int t = (idx >> 9) & 3;
  int c = (idx >> 11) & 1;
  int n = idx >> 12;
  int i = 32 * c + ((L < 16) ? 0 : 8) + h + ((h < 8) ? 0 : 8);
  int o = 16 * t + (L & 15);
  Wf[idx] = (_Float16)W[((size_t)o * DF + i) * NNB + n];
}

// --------------------------------------------------- conv + slice (WMMA) ---
__global__ __launch_bounds__(128) void pl_conv_slice(
    const float*    __restrict__ posOut,   // [B,N,3]
    const unsigned* __restrict__ keysT,
    const float*    __restrict__ barysum,
    const float*    __restrict__ vals,
    const float*    __restrict__ zrow,     // 64 guaranteed-zero floats
    const _Float16* __restrict__ Wf,       // swizzled fragments, 120KB
    const float*    __restrict__ bias,     // [64]
    float*          __restrict__ out)      // [B,64,N]
{
  const int lane  = threadIdx.x & 31;
  const int wave  = threadIdx.x >> 5;
  const int b     = blockIdx.y;
  const int group = blockIdx.x * 4 + wave;   // 4096 groups of 16 rows / batch
  const int row   = lane & 15;               // matrix row M this lane feeds
  const int m     = group * 16 + row;        // output lattice-vertex row
  const int p     = m >> 2;
  const int v     = m & 3;

  const float* pp = posOut + ((size_t)b * N_PTS + p) * 3;
  int rem0[4], rank[4]; float bary[4];
  lattice3(pp[0]*4.f, pp[1]*4.f, pp[2]*4.f, rem0, rank, bary);
  int kb[3];
  #pragma unroll
  for (int i = 0; i < 3; ++i)
    kb[i] = rem0[i] + (rank[i] < 4 - v ? v : v - 4);
  const float myBary = bary[v];

  const unsigned* kt = keysT   + (size_t)b * CAP;
  const float*    bs = barysum + (size_t)b * CAP;
  const float*    vv = vals    + (size_t)b * CAP * DF;

  v8f acc[4] = { {}, {}, {}, {} };
  const int kbaseA = (lane < 16) ? 0 : 8;    // A 16x32 f16 lane layout

  float nbar = 0.f;
  for (int n = 0; n < NNB; ++n) {
    unsigned pk = packKey(kb[0] + c_off[n][0], kb[1] + c_off[n][1], kb[2] + c_off[n][2]);
    int slot = hashFind(kt, pk);
    if (n == 0) nbar = (slot >= 0) ? bs[slot] * myBary : 0.f;

    // Missed keys read from the dedicated zero row: pointer select keeps EXEC
    // full (required for WMMA) and avoids any masking arithmetic.
    const float* g = (slot >= 0) ? (vv + (size_t)slot * DF) : zrow;
    v16h a0, a1;
    #pragma unroll
    for (int h = 0; h < 16; ++h) {
      int K = kbaseA + h + ((h < 8) ? 0 : 8);      // K = base+h (h<8), 16+base+(h-8)
      a0[h] = (_Float16)g[K];
      a1[h] = (_Float16)g[32 + K];
    }

    // B fragments: one contiguous 32B v16h load per (chunk, tile)
    const _Float16* wn = Wf + (size_t)n * 4096 + lane * 16;   // n*2*4*512
    #pragma unroll
    for (int t = 0; t < 4; ++t) {
      v16h b0 = *(const v16h*)(wn + (size_t)t * 512);          // chunk 0
      v16h b1 = *(const v16h*)(wn + (size_t)(4 + t) * 512);    // chunk 1
      acc[t] = __builtin_amdgcn_wmma_f32_16x16x32_f16(false, a0, false, b0,
                                                      (short)0, acc[t], false, false);
      acc[t] = __builtin_amdgcn_wmma_f32_16x16x32_f16(false, a1, false, b1,
                                                      (short)0, acc[t], false, false);
    }
  }

  // ---- slice: wave covers exactly 4 output points (16 rows) ----
  const int mbase = (lane < 16) ? 0 : 8;           // M of acc[t][r] is mbase+r
  float baryRow[8];
  #pragma unroll
  for (int r = 0; r < 8; ++r)
    baryRow[r] = __shfl(myBary, mbase + r, 32);

  const int pbase = (lane < 16) ? 0 : 2;           // local point ids this lane writes
  float norm0 = 0.f, norm1 = 0.f;
  #pragma unroll
  for (int k = 0; k < 4; ++k) {
    norm0 += __shfl(nbar, pbase * 4 + k, 32);
    norm1 += __shfl(nbar, pbase * 4 + 4 + k, 32);
  }
  const float inv0 = 1.0f / (norm0 + 1e-12f);
  const float inv1 = 1.0f / (norm1 + 1e-12f);
  const int pt0 = group * 4 + pbase;

  const int ocol = lane & 15;
  #pragma unroll
  for (int t = 0; t < 4; ++t) {
    float f0 = acc[t][0]*baryRow[0] + acc[t][1]*baryRow[1]
             + acc[t][2]*baryRow[2] + acc[t][3]*baryRow[3];
    float f1 = acc[t][4]*baryRow[4] + acc[t][5]*baryRow[5]
             + acc[t][6]*baryRow[6] + acc[t][7]*baryRow[7];
    int ch = 16 * t + ocol;
    float bb = bias[ch];
    float* o = out + ((size_t)b * DF + ch) * N_PTS;
    o[pt0]     = f0 * inv0 + bb;
    o[pt0 + 1] = f1 * inv1 + bb;
  }
}

// ----------------------------------------------------------------- launch ---
extern "C" void kernel_launch(void* const* d_in, const int* in_sizes, int n_in,
                              void* d_out, int out_size, void* d_ws, size_t ws_size,
                              hipStream_t stream) {
  (void)in_sizes; (void)n_in; (void)out_size; (void)ws_size;
  const float* feat   = (const float*)d_in[0];   // [2,64,16384]
  const float* posIn  = (const float*)d_in[1];   // [2,16384,3]
  const float* posOut = (const float*)d_in[2];   // [2,16384,3]
  const float* W      = (const float*)d_in[3];   // [64,64,15]
  const float* bias   = (const float*)d_in[4];   // [1,64,1]
  float* out = (float*)d_out;                    // [2,64,16384]

  char* ws = (char*)d_ws;
  size_t off = 0;
  _Float16* Wf = (_Float16*)(ws + off); off += (size_t)NNB * 2 * 4 * 32 * 16 * sizeof(_Float16);
  off = (off + 255) & ~(size_t)255;
  unsigned* keysT  = (unsigned*)(ws + off); off += 2ull * CAP * sizeof(unsigned);
  float* barysum   = (float*)(ws + off);    off += 2ull * CAP * sizeof(float);
  float* zrow      = (float*)(ws + off);    off += (size_t)DF * sizeof(float);
  float* vals      = (float*)(ws + off);    off += 2ull * CAP * DF * sizeof(float);

  hipMemsetAsync(keysT,   0xFF, 2ull * CAP * sizeof(unsigned), stream);
  hipMemsetAsync(barysum, 0,    2ull * CAP * sizeof(float),    stream);
  // zrow and vals are contiguous: one memset clears both
  hipMemsetAsync(zrow,    0,    ((size_t)DF + 2ull * CAP * DF) * sizeof(float), stream);

  pl_wswizzle<<<dim3((NNB * 2 * 4 * 32 * 16 + 255) / 256), 256, 0, stream>>>(W, Wf);
  pl_splat<<<dim3(N_PTS / 256, 2), 256, 0, stream>>>(feat, posIn, keysT, barysum, vals);
  pl_conv_slice<<<dim3(1024, 2), 128, 0, stream>>>(posOut, keysT, barysum, vals, zrow, Wf, bias, out);
}
